// GridNet_60739427500683
// MI455X (gfx1250) — matrix operand order repaired and checked by reading, not compiled, not run on hardware
//
#include <hip/hip_runtime.h>
#include <hip/hip_bf16.h>
#include <math.h>

// ---------------------------------------------------------------------------
// GridNet (DirCNN MLP + softmax-weighted 5x5 neighborhood gather) for gfx1250.
// MLP runs on V_WMMA_F32_16X16X32_F16 (f16 inputs, f32 accumulate).
// Weights are pre-packed into WMMA-B fragment-native f16 layout in d_ws.
// Round 2: N-blocked layer loop (NB accumulators per A-fragment load) to cut
// LDS A-fragment traffic 4-5x and chain independent WMMAs.
// ---------------------------------------------------------------------------

typedef __attribute__((ext_vector_type(16))) _Float16     v16h;
typedef __attribute__((ext_vector_type(8)))  float        v8f;
typedef __attribute__((ext_vector_type(8)))  unsigned int v8u;

union AFrag { v16h h; v8u u; };

// Layer geometry: dims 2 -> 64 -> 256 -> 512 -> 256 -> 64 -> 75 (padded to 80)
// K padded to multiples of 32, N padded to multiples of 16.
static constexpr int   KT[6]    = {1, 2, 8, 16, 8, 2};
static constexpr int   NT[6]    = {4, 16, 32, 16, 4, 5};
static constexpr int   DIN_[6]  = {2, 64, 256, 512, 256, 64};
static constexpr int   DOUT_[6] = {64, 256, 512, 256, 64, 75};
// cumulative fragment offsets (in halves); each fragment is 512 halves
static constexpr size_t WOFF[6] = {0, 2048, 18432, 149504, 280576, 296960};
// total packed halves = 302080 -> 604160 bytes of d_ws

static constexpr int GP1   = 2052;   // grid_pos dim1 = 2048 + 4
static constexpr int WAVES = 2;      // waves per block

// ---------------------------------------------------------------------------
// Pre-pack: fp32 weight [din x dout] -> f16 fragments.
// Fragment (kt, nt) holds B-tile K = kt*32..+31, N = nt*16..+15 with per-lane
// layout: lanes 0-15: K = kt*32 + i (i = 0..15), N = nt*16 + lane
//         lanes 16-31: K = kt*32 + 16 + i,       N = nt*16 + lane - 16
// Packed order: dst[frag*512 + lane*16 + i]  (main kernel reads one v16h/lane)
// ---------------------------------------------------------------------------
__global__ void gridnet_prepack(const float* __restrict__ W,
                                _Float16* __restrict__ dst,
                                int din, int dout, int Nt, int total) {
    int t = blockIdx.x * blockDim.x + threadIdx.x;
    if (t >= total) return;
    int frag = t >> 9;          // / 512
    int lane = (t >> 4) & 31;
    int i    = t & 15;
    int kt = frag / Nt;
    int nt = frag - kt * Nt;
    int k  = kt * 32 + ((lane < 16) ? i : (16 + i));
    int n  = nt * 16 + (lane & 15);
    float v = (k < din && n < dout) ? W[(size_t)k * dout + n] : 0.0f;
    dst[t] = (_Float16)v;
}

// ---------------------------------------------------------------------------
// One MLP layer for a 16-row tile owned by this wave, N-blocked by NB.
//   inb : row-major [16][ldin]  f16 activations (LDS)
//   outb: row-major [16][ldout] f16 activations (LDS)
// A-fragment 16-bit layout (contiguous K, matching packed B):
//   lane<16:  VGPR r holds K = 2r,2r+1
//   lane>=16: VGPR r holds K = 2r+16,2r+17
// C/D layout: VGPR r -> row M = r + 8*(lane>=16), col N = lane&15.
// Requires Nt % NB == 0.
// ---------------------------------------------------------------------------
template <int NB>
__device__ __forceinline__ void mlp_layer(const _Float16* __restrict__ wpack,
                                          const float* __restrict__ bias,
                                          const _Float16* __restrict__ inb, int ldin,
                                          _Float16* __restrict__ outb, int ldout,
                                          int Kt, int Nt, int dout, bool relu,
                                          int lane) {
    const int mrow = lane & 15;
    const int hi   = lane >> 4;
    const unsigned int* row32 = (const unsigned int*)(inb + mrow * ldin);
    for (int nt0 = 0; nt0 < Nt; nt0 += NB) {
        v8f acc[NB];
#pragma unroll
        for (int q = 0; q < NB; ++q) {
            const int col = (nt0 + q) * 16 + mrow;
            float bb = (col < dout) ? bias[col] : 0.0f;
#pragma unroll
            for (int r = 0; r < 8; ++r) acc[q][r] = bb;
        }
        for (int kt = 0; kt < Kt; ++kt) {
            // Load A-fragment once, use it for NB independent WMMAs.
            AFrag a;
#pragma unroll
            for (int r = 0; r < 8; ++r) {
                int k = kt * 32 + 2 * r + 16 * hi;
                a.u[r] = row32[k >> 1];
            }
            const v16h* wrow = (const v16h*)wpack + ((size_t)(kt * Nt + nt0) * 32 + lane);
#pragma unroll
            for (int q = 0; q < NB; ++q) {
                v16h bf = wrow[q * 32];
                acc[q] = __builtin_amdgcn_wmma_f32_16x16x32_f16(
                    false, a.h, false, bf, (short)0, acc[q], false, false);
            }
        }
#pragma unroll
        for (int q = 0; q < NB; ++q) {
#pragma unroll
            for (int r = 0; r < 8; ++r) {
                float v = acc[q][r];
                if (relu) v = fmaxf(v, 0.0f);
                outb[(r + 8 * hi) * ldout + (nt0 + q) * 16 + mrow] = (_Float16)v;
            }
        }
    }
}

// ---------------------------------------------------------------------------
// Main kernel: one wave per 16-row batch tile. Per-wave private LDS region:
// buf0 = 16x512 halves, buf1 = 16x256 halves (ping-pong; no cross-wave sync:
// same-wave LDS RAW is ordered by compiler-inserted s_wait_dscnt).
// ---------------------------------------------------------------------------
__global__ void __launch_bounds__(WAVES * 32)
gridnet_main(const float* __restrict__ pos,
             const float* __restrict__ grid,
             const _Float16* __restrict__ wp,
             const float* __restrict__ b1, const float* __restrict__ b2,
             const float* __restrict__ b3, const float* __restrict__ b4,
             const float* __restrict__ b5, const float* __restrict__ b6,
             float* __restrict__ out) {
    extern __shared__ _Float16 smem[];
    const int lane = threadIdx.x & 31;
    const int wid  = threadIdx.x >> 5;
    const int tile = blockIdx.x * WAVES + wid;
    const int row0 = tile * 16;

    _Float16* buf0 = smem + (size_t)wid * (16 * (512 + 256));
    _Float16* buf1 = buf0 + 16 * 512;

    const int mrow = lane & 15;
    const int hi   = lane >> 4;

    // ---- Layer 1: input [16x2] built directly in A-fragment registers -----
    // lane<16, i=0 -> K=0 (theta); i=1 -> K=1 (phi). lanes>=16 hold K>=16 => 0,
    // matching zero-padded weight rows k >= 2.
    AFrag a0;
#pragma unroll
    for (int r = 0; r < 8; ++r) a0.u[r] = 0u;
    if (hi == 0) {
        int gi = row0 + mrow;
        a0.h[0] = (_Float16)pos[2 * gi];
        a0.h[1] = (_Float16)pos[2 * gi + 1];
    }
    {
        v8f acc[4];
#pragma unroll
        for (int q = 0; q < 4; ++q) {
            float bb = b1[q * 16 + mrow];
#pragma unroll
            for (int r = 0; r < 8; ++r) acc[q][r] = bb;
        }
        const v16h* wrow = (const v16h*)(wp + WOFF[0]) + lane;
#pragma unroll
        for (int q = 0; q < 4; ++q) {
            v16h bf = wrow[q * 32];
            acc[q] = __builtin_amdgcn_wmma_f32_16x16x32_f16(
                false, a0.h, false, bf, (short)0, acc[q], false, false);
        }
#pragma unroll
        for (int q = 0; q < 4; ++q) {
#pragma unroll
            for (int r = 0; r < 8; ++r) {
                float v = fmaxf(acc[q][r], 0.0f);
                buf0[(r + 8 * hi) * 512 + q * 16 + mrow] = (_Float16)v;
            }
        }
    }

    // ---- Layers 2..6 (NB = 4, logits layer NB = 5) -------------------------
    mlp_layer<4>(wp + WOFF[1], b2, buf0, 512, buf1, 256, KT[1], NT[1], DOUT_[1], true,  lane);
    mlp_layer<4>(wp + WOFF[2], b3, buf1, 256, buf0, 512, KT[2], NT[2], DOUT_[2], true,  lane);
    mlp_layer<4>(wp + WOFF[3], b4, buf0, 512, buf1, 256, KT[3], NT[3], DOUT_[3], true,  lane);
    mlp_layer<4>(wp + WOFF[4], b5, buf1, 256, buf0, 512, KT[4], NT[4], DOUT_[4], true,  lane);
    mlp_layer<5>(wp + WOFF[5], b6, buf0, 512, buf1, 256, KT[5], NT[5], DOUT_[5], false, lane);
    // logits (75 valid of 80) now in buf1, row-major ld=256, f16

    // ---- Softmax over 75 logits + 3x5x5 gather -----------------------------
    // 2 lanes per batch row: lane = row + 16*half; each handles ~13 of the 25
    // grid positions, partials combined with a cross-half shuffle.
    const int row  = mrow;
    const int half = hi;
    const _Float16* lg = buf1 + row * 256;

    float m = -3.0e38f;
    for (int j = 0; j < 75; ++j) m = fmaxf(m, (float)lg[j]);
    float ssum = 0.0f;
    for (int j = 0; j < 75; ++j) ssum += __expf((float)lg[j] - m);
    float inv = 1.0f / ssum;

    const int gi = row0 + row;
    const float PI_F     = 3.14159265358979323846f;
    const float INV_PI   = 0.31830988618379067154f;
    const float INV_2PI  = 0.15915494309189533577f;
    float theta = pos[2 * gi];
    float phi   = pos[2 * gi + 1];
    float gx = theta * INV_PI * 1023.0f;                 // (GS0-1)
    float gy = (phi + PI_F) * INV_2PI * 2047.0f;         // (GS1-1)
    int tlx = (int)gx;
    int tly = (int)gy;

    float s0 = 0.0f, s1 = 0.0f, s2 = 0.0f;
    for (int p = half; p < 25; p += 2) {
        int i = p / 5;
        int j = p - 5 * i;
        const float* g = grid + ((size_t)(tlx + i) * GP1 + (tly + j)) * 3;
        float w0 = __expf((float)lg[p]      - m) * inv;
        float w1 = __expf((float)lg[25 + p] - m) * inv;
        float w2 = __expf((float)lg[50 + p] - m) * inv;
        s0 += w0 * g[0];
        s1 += w1 * g[1];
        s2 += w2 * g[2];
    }
    s0 += __shfl_xor(s0, 16);
    s1 += __shfl_xor(s1, 16);
    s2 += __shfl_xor(s2, 16);

    if (half == 0) {
        float rv[3] = {s0, s1, s2};
#pragma unroll
        for (int c = 0; c < 3; ++c) {
            float v = 1.0f / (1.0f + __expf(-rv[c]));    // sigmoid
            v = (v > 0.1f) ? v : 0.0f;                   // threshold
            out[(size_t)gi * 3 + c] = v * 255.0f;
        }
    }
}

// ---------------------------------------------------------------------------
// Host launcher
// ---------------------------------------------------------------------------
extern "C" void kernel_launch(void* const* d_in, const int* in_sizes, int n_in,
                              void* d_out, int out_size, void* d_ws, size_t ws_size,
                              hipStream_t stream) {
    const float* pos  = (const float*)d_in[0];
    const float* grid = (const float*)d_in[1];
    const float* W[6]  = {(const float*)d_in[2],  (const float*)d_in[4],
                          (const float*)d_in[6],  (const float*)d_in[8],
                          (const float*)d_in[10], (const float*)d_in[12]};
    const float* Bv[6] = {(const float*)d_in[3],  (const float*)d_in[5],
                          (const float*)d_in[7],  (const float*)d_in[9],
                          (const float*)d_in[11], (const float*)d_in[13]};
    float* out = (float*)d_out;
    _Float16* wp = (_Float16*)d_ws;

    // Pack weights into WMMA fragment layout (f16) in workspace. Deterministic
    // and re-done every call (harness poisons d_ws).
    for (int l = 0; l < 6; ++l) {
        int total = KT[l] * NT[l] * 512;
        int blocks = (total + 255) / 256;
        gridnet_prepack<<<blocks, 256, 0, stream>>>(W[l], wp + WOFF[l],
                                                    DIN_[l], DOUT_[l], NT[l], total);
    }

    int B = in_sizes[0] / 2;          // 262144
    int nTiles = B / 16;              // 16384 row tiles, one per wave
    int nBlocks = nTiles / WAVES;     // 8192 blocks of 2 waves
    size_t shmem = (size_t)WAVES * 16 * (512 + 256) * sizeof(_Float16); // 48 KB
    gridnet_main<<<nBlocks, WAVES * 32, shmem, stream>>>(
        pos, grid, wp, Bv[0], Bv[1], Bv[2], Bv[3], Bv[4], Bv[5], out);
}
